// ClassifierLSTMDeltas_5446018531506
// MI455X (gfx1250) — matrix operand order repaired and checked by reading, not compiled, not run on hardware
//
#include <hip/hip_runtime.h>
#include <hip/hip_bf16.h>

// ---------------- CDNA5 WMMA types / helpers ----------------
typedef __attribute__((ext_vector_type(16))) _Float16 v16h;
typedef __attribute__((ext_vector_type(8)))  _Float16 v8h;
typedef __attribute__((ext_vector_type(4)))  _Float16 v4h;
typedef __attribute__((ext_vector_type(8)))  float    v8f;

#define WMMA_F16(a, b, c) \
  __builtin_amdgcn_wmma_f32_16x16x32_f16(false, (a), false, (b), (short)0, (c), false, false)

// Fragment load: lane reads 16 *contiguous* halves at base + (lane&15)*lda +
// (lane>=16 ? 16 : 0).  Tiles are stored in fragment order (see producers), so
// this lowers to exactly two ds_load_b128 per fragment.
__device__ __forceinline__ v16h frag_ld(const _Float16* base, int lda) {
  const int lane = threadIdx.x & 31;
  const _Float16* p = base + (lane & 15) * lda + ((lane >> 4) << 4);
  union { v16h f; v8h h[2]; } u;
  u.h[0] = *(const v8h*)(p);
  u.h[1] = *(const v8h*)(p + 8);
  return u.f;
}

__device__ __forceinline__ v8f zero8() {
  v8f z;
#pragma unroll
  for (int i = 0; i < 8; ++i) z[i] = 0.0f;
  return z;
}

__device__ __forceinline__ float gelu_f(float x) {
  return 0.5f * x * (1.0f + erff(x * 0.70710678118654752f));
}
__device__ __forceinline__ float sigm_f(float x) { return 1.0f / (1.0f + expf(-x)); }

// A-tile 8-block mapping: identity (natural k order), matching the B tiles'
// contiguous k = 16h + s fragment order.
__device__ __forceinline__ int ablk(int blk) {
  const int bmap[4] = {0, 1, 2, 3};
  return bmap[blk];
}

// ---------------- problem constants ----------------
#define BATCH 4096
#define TT    31
#define CH    768
#define BD    128
#define AUG   384
#define HID   64
#define OUTD  12
#define NTROWS (BATCH * TT)          // 126976 = 64 * 1984
#define WIN_L 10
#define WIN_R 21
#define WINW  11
#define EMA_ALPHA 0.3f
#define LN_EPS 1e-5f

// =====================================================================
// K1: EMA smoothing over T, output f16 x_smooth  (B blocks x 256 thr)
// =====================================================================
__global__ __launch_bounds__(256) void ema_kernel(const float* __restrict__ x,
                                                  _Float16* __restrict__ xs) {
  const long long base = (long long)blockIdx.x * TT * CH;
  for (int c = threadIdx.x; c < CH; c += 256) {
    float s = x[base + c];
    xs[base + c] = (_Float16)s;
#pragma unroll 1
    for (int t = 1; t < TT; ++t) {
      float v = x[base + (long long)t * CH + c];
      s += EMA_ALPHA * (v - s);
      xs[base + (long long)t * CH + c] = (_Float16)s;
    }
  }
}

// =====================================================================
// K2: linear_logits = mean_t( xs[:,10:21] @ lin1_w ) + lin1_b  (B,12)
// =====================================================================
__global__ __launch_bounds__(256) void linlog_kernel(const _Float16* __restrict__ xs,
                                                     const float* __restrict__ w,
                                                     const float* __restrict__ bias,
                                                     float* __restrict__ out) {
  __shared__ float red[256][OUTD];
  const int b = blockIdx.x;
  float acc[OUTD];
#pragma unroll
  for (int o = 0; o < OUTD; ++o) acc[o] = 0.0f;
  for (int c = threadIdx.x; c < CH; c += 256) {
    float wv[OUTD];
#pragma unroll
    for (int o = 0; o < OUTD; ++o) wv[o] = w[c * OUTD + o];
#pragma unroll 1
    for (int t = WIN_L; t < WIN_R; ++t) {
      float xv = (float)xs[((long long)b * TT + t) * CH + c];
#pragma unroll
      for (int o = 0; o < OUTD; ++o) acc[o] += xv * wv[o];
    }
  }
#pragma unroll
  for (int o = 0; o < OUTD; ++o) red[threadIdx.x][o] = acc[o];
  __syncthreads();
  if (threadIdx.x < OUTD) {
    float s = 0.0f;
    for (int j = 0; j < 256; ++j) s += red[j][threadIdx.x];
    out[(long long)b * OUTD + threadIdx.x] = s * (1.0f / (float)WINW) + bias[threadIdx.x];
  }
}

// =====================================================================
// K3: fused triple GEMM (cls/delta/acc streams built on-the-fly from
//     x_smooth) + GELU + LayerNorm  -> x_aug f16 (NT x 384)
//     Tiles stored fragment-major; 12 WMMAs per K-chunk per wave.
// =====================================================================
struct K3Ld {
  _Float16 As[3][64][40];      // rows x K-chunk, natural k order
  _Float16 Bs[3][8][16][40];   // [ntile][col][k] fragment-major
};
struct K3Ep { float St[64][132]; float mu[64]; float rs[64]; };
union K3Sm { K3Ld ld; K3Ep ep; };

__global__ __launch_bounds__(256) void triple_gemm_ln_kernel(
    const _Float16* __restrict__ xs,
    const float* __restrict__ w0, const float* __restrict__ b0,
    const float* __restrict__ w1, const float* __restrict__ b1,
    const float* __restrict__ w2, const float* __restrict__ b2,
    const float* __restrict__ g0, const float* __restrict__ be0,
    const float* __restrict__ g1, const float* __restrict__ be1,
    const float* __restrict__ g2, const float* __restrict__ be2,
    _Float16* __restrict__ xaug) {
  __shared__ K3Sm sm;
  const int row0 = blockIdx.x * 64;
  const int tid  = threadIdx.x;
  const int wid  = tid >> 5;
  const int lane = tid & 31;
  const int mt   = wid & 3;   // m-tile (16 rows each)
  const int nh   = wid >> 2;  // n-half (64 cols each)

  v8f acc[3][4];
#pragma unroll
  for (int s = 0; s < 3; ++s)
#pragma unroll
    for (int j = 0; j < 4; ++j) acc[s][j] = zero8();

#pragma unroll 1
  for (int k0 = 0; k0 < CH; k0 += 32) {
    __syncthreads();
    // ---- A tiles: one 8-half block per thread (64 rows x 4 blocks) ----
    {
      const int r = tid >> 2, blk = tid & 3;
      const long long n = row0 + r;
      const int t = (int)(n % TT);
      const long long idx = n * CH + k0 + blk * 8;
      const v8h p0 = *(const v8h*)(xs + idx);
      v8h d, a;
      if (t == 0) {
        const v8h p1 = *(const v8h*)(xs + idx + CH);
        const v8h p2 = *(const v8h*)(xs + idx + 2 * CH);
#pragma unroll
        for (int i = 0; i < 8; ++i) {
          const float v0 = (float)p0[i], f1 = (float)p1[i], f2 = (float)p2[i];
          d[i] = (_Float16)(v0 - f1);
          a[i] = (_Float16)(v0 - 2.0f * f1 + f2);
        }
      } else if (t == 1) {
        const v8h m1 = *(const v8h*)(xs + idx - CH);
#pragma unroll
        for (int i = 0; i < 8; ++i) {
          const float dv = (float)p0[i] - (float)m1[i];
          d[i] = (_Float16)dv;
          a[i] = (_Float16)(2.0f * dv);
        }
      } else {
        const v8h m1 = *(const v8h*)(xs + idx - CH);
        const v8h m2 = *(const v8h*)(xs + idx - 2 * CH);
#pragma unroll
        for (int i = 0; i < 8; ++i) {
          const float v0 = (float)p0[i], f1 = (float)m1[i], f2 = (float)m2[i];
          d[i] = (_Float16)(v0 - f1);
          a[i] = (_Float16)(v0 - 2.0f * f1 + f2);
        }
      }
      const int bp = ablk(blk) * 8;
      *(v8h*)&sm.ld.As[0][r][bp] = p0;
      *(v8h*)&sm.ld.As[1][r][bp] = d;
      *(v8h*)&sm.ld.As[2][r][bp] = a;
    }
    // ---- B tiles: 4 K-values per thread, fragment-major store ----
    for (int e = tid; e < 1024; e += 256) {
      const int c  = e & 127;
      const int kb = (e >> 7) * 4;
      const int jt = c >> 4, col = c & 15;
      v4h q0, q1, q2;
#pragma unroll
      for (int i = 0; i < 4; ++i) {
        const long long widx = (long long)(k0 + kb + i) * BD + c;
        q0[i] = (_Float16)w0[widx];
        q1[i] = (_Float16)w1[widx];
        q2[i] = (_Float16)w2[widx];
      }
      *(v4h*)&sm.ld.Bs[0][jt][col][kb] = q0;
      *(v4h*)&sm.ld.Bs[1][jt][col][kb] = q1;
      *(v4h*)&sm.ld.Bs[2][jt][col][kb] = q2;
      if (k0 + 32 < CH)  // global_prefetch for next chunk
        __builtin_prefetch(&w0[(long long)(k0 + 32 + kb) * BD + c], 0, 0);
    }
    __syncthreads();
#pragma unroll
    for (int s = 0; s < 3; ++s) {
      const v16h af = frag_ld(&sm.ld.As[s][mt * 16][0], 40);
#pragma unroll
      for (int j = 0; j < 4; ++j) {
        const v16h bf = frag_ld(&sm.ld.Bs[s][nh * 4 + j][0][0], 40);
        acc[s][j] = WMMA_F16(af, bf, acc[s][j]);
      }
    }
  }

  // Epilogue per stream: bias + GELU -> LDS stage -> LayerNorm(128) -> f16
  const float* BIAS[3] = {b0, b1, b2};
  const float* LG[3]   = {g0, g1, g2};
  const float* LB[3]   = {be0, be1, be2};
#pragma unroll 1
  for (int s = 0; s < 3; ++s) {
    __syncthreads();
#pragma unroll
    for (int j = 0; j < 4; ++j)
#pragma unroll
      for (int r = 0; r < 8; ++r) {
        const int row = mt * 16 + ((lane & 16) ? 8 + r : r);
        const int col = nh * 64 + j * 16 + (lane & 15);
        sm.ep.St[row][col] = gelu_f(acc[s][j][r] + BIAS[s][col]);
      }
    __syncthreads();
    if (tid < 64) {
      float m = 0.0f;
      for (int c = 0; c < BD; ++c) m += sm.ep.St[tid][c];
      m *= (1.0f / BD);
      float var = 0.0f;
      for (int c = 0; c < BD; ++c) {
        const float dv = sm.ep.St[tid][c] - m;
        var += dv * dv;
      }
      sm.ep.mu[tid] = m;
      sm.ep.rs[tid] = rsqrtf(var * (1.0f / BD) + LN_EPS);
    }
    __syncthreads();
    for (int e = tid; e < 64 * 128; e += 256) {
      const int row = e >> 7, col = e & 127;
      const float y = (sm.ep.St[row][col] - sm.ep.mu[row]) * sm.ep.rs[row] * LG[s][col] + LB[s][col];
      xaug[(long long)(row0 + row) * AUG + s * BD + col] = (_Float16)y;
    }
  }
}

// =====================================================================
// K4/K6: generic f16-A x f32-weight WMMA GEMM + bias (+optional GELU)
// =====================================================================
__global__ __launch_bounds__(256) void gemm_bias_act_kernel(
    const _Float16* __restrict__ A, const float* __restrict__ Bw,
    const float* __restrict__ bias, float* __restrict__ Cf,
    int K, int N, int act) {
  __shared__ _Float16 As[64][40];
  __shared__ _Float16 Bs[8][16][40];
  const int row0 = blockIdx.x * 64;
  const int col0 = blockIdx.y * 128;
  const int tid  = threadIdx.x;
  const int wid  = tid >> 5;
  const int lane = tid & 31;
  const int mt   = wid & 3;
  const int nh   = wid >> 2;

  v8f acc[4];
#pragma unroll
  for (int j = 0; j < 4; ++j) acc[j] = zero8();

#pragma unroll 1
  for (int k0 = 0; k0 < K; k0 += 32) {
    __syncthreads();
    {  // A: one 8-half block per thread, natural-order store (b128 in/out)
      const int r = tid >> 2, blk = tid & 3;
      const v8h p = *(const v8h*)(A + (long long)(row0 + r) * K + k0 + blk * 8);
      *(v8h*)&As[r][ablk(blk) * 8] = p;
    }
    for (int e = tid; e < 1024; e += 256) {  // B: 4 K-values per thread
      const int c  = e & 127;
      const int kb = (e >> 7) * 4;
      const int jt = c >> 4, col = c & 15;
      v4h q;
#pragma unroll
      for (int i = 0; i < 4; ++i)
        q[i] = (_Float16)Bw[(long long)(k0 + kb + i) * N + col0 + c];
      *(v4h*)&Bs[jt][col][kb] = q;
      if (k0 + 32 < K)
        __builtin_prefetch(&Bw[(long long)(k0 + 32 + kb) * N + col0 + c], 0, 0);
    }
    __syncthreads();
    const v16h af = frag_ld(&As[mt * 16][0], 40);
#pragma unroll
    for (int j = 0; j < 4; ++j) {
      const v16h bf = frag_ld(&Bs[nh * 4 + j][0][0], 40);
      acc[j] = WMMA_F16(af, bf, acc[j]);
    }
  }

#pragma unroll
  for (int j = 0; j < 4; ++j)
#pragma unroll
    for (int r = 0; r < 8; ++r) {
      const int row = row0 + mt * 16 + ((lane & 16) ? 8 + r : r);
      const int col = col0 + nh * 64 + j * 16 + (lane & 15);
      float v = acc[j][r] + bias[col];
      if (act) v = gelu_f(v);
      Cf[(long long)row * N + col] = v;
    }
}

// =====================================================================
// K5: mean-center over T per (b, feature), f32 -> f16
// =====================================================================
__global__ __launch_bounds__(256) void center_kernel(const float* __restrict__ xin,
                                                     _Float16* __restrict__ xout) {
  const long long base = (long long)blockIdx.x * TT * 256 + threadIdx.x;
  float s = 0.0f;
#pragma unroll 1
  for (int t = 0; t < TT; ++t) s += xin[base + (long long)t * 256];
  s *= (1.0f / (float)TT);
#pragma unroll 1
  for (int t = 0; t < TT; ++t)
    xout[base + (long long)t * 256] = (_Float16)(xin[base + (long long)t * 256] - s);
}

// =====================================================================
// K_rec: LSTM recurrence, one direction; 16 batch rows per block.
//        w_hh fragment-major in LDS; h stored in natural k order so both
//        A chunks are 2x ds_load_b128. 4 WMMAs/wave/step.
// =====================================================================
__global__ __launch_bounds__(256) void lstm_rec_kernel(
    const float* __restrict__ pre,   // NT x 256, natural t order
    const float* __restrict__ whh,   // 64 x 256 f32
    float* __restrict__ win,         // B x 11 x 128
    int dir) {
  __shared__ _Float16 Wh[16][16][72];  // [ntile][col][k0..63]
  __shared__ _Float16 Hh[16][80];      // row x {chunk0: 0..39, chunk1: 40..79}
  __shared__ float    Cst[16][64];
  __shared__ float    Gs[16][260];

  const int tid  = threadIdx.x;
  const int wid  = tid >> 5;
  const int lane = tid & 31;
  const int b0   = blockIdx.x * 16;

  // w_hh -> LDS fragment-major (4 K-values per thread -> ds_store_b64)
  for (int e = tid; e < 16 * 16 * 16; e += 256) {
    const int jt = e >> 8, col = (e >> 4) & 15, kb = (e & 15) * 4;
    v4h q;
#pragma unroll
    for (int i = 0; i < 4; ++i)
      q[i] = (_Float16)whh[(long long)(kb + i) * 256 + jt * 16 + col];
    *(v4h*)&Wh[jt][col][kb] = q;
  }
  for (int e = tid; e < 16 * 80; e += 256) Hh[e >> 6 ? (e / 80) : 0][e % 80] = (_Float16)0.0f;
  for (int e = tid; e < 16 * 64; e += 256) Cst[e >> 6][e & 63] = 0.0f;
  __syncthreads();

#pragma unroll 1
  for (int s = 0; s < TT; ++s) {
    const int t = dir ? (TT - 1 - s) : s;
    // init accumulators from the precomputed input projection
    v8f acc[2];
#pragma unroll
    for (int j = 0; j < 2; ++j) {
#pragma unroll
      for (int r = 0; r < 8; ++r) {
        const int row = (lane & 16) ? 8 + r : r;
        const int col = 32 * wid + 16 * j + (lane & 15);
        acc[j][r] = pre[((long long)(b0 + row) * TT + t) * 256 + col];
      }
    }
    const v16h a0 = frag_ld(&Hh[0][0], 80);   // K 0..31
    const v16h a1 = frag_ld(&Hh[0][40], 80);  // K 32..63
#pragma unroll
    for (int j = 0; j < 2; ++j) {
      const int jt = wid * 2 + j;
      const v16h bf0 = frag_ld(&Wh[jt][0][0], 72);
      acc[j] = WMMA_F16(a0, bf0, acc[j]);
      const v16h bf1 = frag_ld(&Wh[jt][0][32], 72);
      acc[j] = WMMA_F16(a1, bf1, acc[j]);
    }
#pragma unroll
    for (int j = 0; j < 2; ++j)
#pragma unroll
      for (int r = 0; r < 8; ++r) {
        const int row = (lane & 16) ? 8 + r : r;
        Gs[row][32 * wid + 16 * j + (lane & 15)] = acc[j][r];
      }
    __syncthreads();
    // gate nonlinearity + state update (i|f|g|o column blocks)
    for (int e = tid; e < 16 * 64; e += 256) {
      const int r = e >> 6, j = e & 63;
      const float ig = sigm_f(Gs[r][j]);
      const float fg = sigm_f(Gs[r][64 + j]);
      const float gg = tanhf(Gs[r][128 + j]);
      const float og = sigm_f(Gs[r][192 + j]);
      const float c  = fg * Cst[r][j] + ig * gg;
      const float h  = og * tanhf(c);
      Cst[r][j] = c;
      const int chunk = j >> 5, kk = j & 31;
      Hh[r][chunk * 40 + ablk(kk >> 3) * 8 + (kk & 7)] = (_Float16)h;
      if (t >= WIN_L && t < WIN_R)
        win[((long long)(b0 + r) * WINW + (t - WIN_L)) * 128 + dir * HID + j] = h;
    }
    __syncthreads();
  }
}

// =====================================================================
// K7: attention over window + output heads + gating. One block per b.
// =====================================================================
__global__ __launch_bounds__(128) void attn_final_kernel(
    const float* __restrict__ win, const float* __restrict__ attn_w,
    const float* __restrict__ attn_b, const float* __restrict__ attn_temp,
    const float* __restrict__ lin2_w, const float* __restrict__ lin2_b,
    const float* __restrict__ linlog, const float* __restrict__ gate,
    float* __restrict__ out) {
  __shared__ float Ws[WINW][132];
  __shared__ float sc[WINW];
  __shared__ float att[132];
  const int b = blockIdx.x;
  const int tid = threadIdx.x;

  for (int e = tid; e < WINW * 128; e += 128)
    Ws[e >> 7][e & 127] = win[((long long)b * WINW) * 128 + e];
  __syncthreads();
  if (tid < WINW) {
    float s = 0.0f;
    for (int j = 0; j < 128; ++j) s += Ws[tid][j] * attn_w[j];
    const float ti = attn_temp[0];
    const float sp = (ti > 20.0f) ? ti : log1pf(expf(ti));
    sc[tid] = (s + attn_b[0]) / (sp + 0.001f);
  }
  __syncthreads();
  if (tid == 0) {
    float m = sc[0];
    for (int t = 1; t < WINW; ++t) m = fmaxf(m, sc[t]);
    float sum = 0.0f;
    for (int t = 0; t < WINW; ++t) { sc[t] = expf(sc[t] - m); sum += sc[t]; }
    const float inv = 1.0f / sum;
    for (int t = 0; t < WINW; ++t) sc[t] *= inv;
  }
  __syncthreads();
  float a = 0.0f;
#pragma unroll
  for (int t = 0; t < WINW; ++t) a += sc[t] * Ws[t][tid];
  att[tid] = a;
  out[(long long)BATCH * OUTD + (long long)b * 128 + tid] = a;  // attended
  __syncthreads();
  if (tid < OUTD) {
    float s = lin2_b[tid];
    for (int j = 0; j < 128; ++j) s += att[j] * lin2_w[j * OUTD + tid];
    const float g = sigm_f(gate[0]);
    const float lin = linlog[(long long)b * OUTD + tid];
    out[(long long)b * OUTD + tid] = lin + g * (s - lin);
  }
}

// =====================================================================
extern "C" void kernel_launch(void* const* d_in, const int* in_sizes, int n_in,
                              void* d_out, int out_size, void* d_ws, size_t ws_size,
                              hipStream_t stream) {
  const float* x         = (const float*)d_in[0];
  const float* cls_w     = (const float*)d_in[1];
  const float* cls_b     = (const float*)d_in[2];
  const float* delta_w   = (const float*)d_in[3];
  const float* delta_b   = (const float*)d_in[4];
  const float* acc_w     = (const float*)d_in[5];
  const float* acc_b     = (const float*)d_in[6];
  const float* cls_ln_g  = (const float*)d_in[7];
  const float* cls_ln_b  = (const float*)d_in[8];
  const float* delta_ln_g= (const float*)d_in[9];
  const float* delta_ln_b= (const float*)d_in[10];
  const float* acc_ln_g  = (const float*)d_in[11];
  const float* acc_ln_b  = (const float*)d_in[12];
  const float* lin0_w    = (const float*)d_in[13];
  const float* lin0_b    = (const float*)d_in[14];
  const float* gate      = (const float*)d_in[15];
  const float* attn_w    = (const float*)d_in[16];
  const float* attn_b    = (const float*)d_in[17];
  const float* attn_temp = (const float*)d_in[18];
  const float* lin1_w    = (const float*)d_in[19];
  const float* lin1_b    = (const float*)d_in[20];
  const float* lin2_w    = (const float*)d_in[21];
  const float* lin2_b    = (const float*)d_in[22];
  const float* wih_f     = (const float*)d_in[23];
  const float* whh_f     = (const float*)d_in[24];
  const float* b_f       = (const float*)d_in[25];
  const float* wih_r     = (const float*)d_in[26];
  const float* whh_r     = (const float*)d_in[27];
  const float* b_r       = (const float*)d_in[28];
  float* out = (float*)d_out;

  // ---- workspace layout (region reuse; all offsets 256B aligned) ----
  char* ws = (char*)d_ws;
  const size_t off_xs   = 0;                                   // f16 NT*768
  const size_t off_xaug = off_xs   + (size_t)NTROWS * CH * 2;  // f16 NT*384
  const size_t off_xlg  = off_xaug + (size_t)NTROWS * AUG * 2; // f32 NT*256
  const size_t off_xlh  = off_xlg  + (size_t)NTROWS * 256 * 4; // f16 NT*256
  const size_t off_win  = off_xlh  + (size_t)NTROWS * 256 * 2; // f32 B*11*128
  const size_t off_ll   = off_win  + (size_t)BATCH * WINW * 128 * 4; // f32 B*12
  _Float16* xs_h   = (_Float16*)(ws + off_xs);
  _Float16* xaug_h = (_Float16*)(ws + off_xaug);
  float*    xlg    = (float*)(ws + off_xlg);
  _Float16* xlh    = (_Float16*)(ws + off_xlh);
  float*    win    = (float*)(ws + off_win);
  float*    linlog = (float*)(ws + off_ll);
  // pre-gate buffers alias xs/xaug regions (dead by the time they're used)
  float* preF = (float*)(ws + 0);
  float* preR = (float*)(ws + (size_t)NTROWS * 256 * 4);

  // ---- pipeline ----
  ema_kernel<<<BATCH, 256, 0, stream>>>(x, xs_h);
  linlog_kernel<<<BATCH, 256, 0, stream>>>(xs_h, lin1_w, lin1_b, linlog);
  triple_gemm_ln_kernel<<<NTROWS / 64, 256, 0, stream>>>(
      xs_h, cls_w, cls_b, delta_w, delta_b, acc_w, acc_b,
      cls_ln_g, cls_ln_b, delta_ln_g, delta_ln_b, acc_ln_g, acc_ln_b, xaug_h);
  gemm_bias_act_kernel<<<dim3(NTROWS / 64, 2), 256, 0, stream>>>(
      xaug_h, lin0_w, lin0_b, xlg, AUG, 256, /*gelu=*/1);
  center_kernel<<<BATCH, 256, 0, stream>>>(xlg, xlh);
  gemm_bias_act_kernel<<<dim3(NTROWS / 64, 2), 256, 0, stream>>>(
      xlh, wih_f, b_f, preF, 256, 256, /*act=*/0);
  gemm_bias_act_kernel<<<dim3(NTROWS / 64, 2), 256, 0, stream>>>(
      xlh, wih_r, b_r, preR, 256, 256, /*act=*/0);
  lstm_rec_kernel<<<BATCH / 16, 256, 0, stream>>>(preF, whh_f, win, 0);
  lstm_rec_kernel<<<BATCH / 16, 256, 0, stream>>>(preR, whh_r, win, 1);
  attn_final_kernel<<<BATCH, 128, 0, stream>>>(
      win, attn_w, attn_b, attn_temp, lin2_w, lin2_b, linlog, gate, out);
  (void)in_sizes; (void)n_in; (void)out_size; (void)ws_size;
}
